// LocalGreedySNN_62723702391157
// MI455X (gfx1250) — matrix operand order repaired and checked by reading, not compiled, not run on hardware
//
#include <hip/hip_runtime.h>
#include <hip/hip_bf16.h>

// ---------------------------------------------------------------------------
// Spiking MLP (784 -> 1024 -> 1024 -> 10), T=32, LIF tau=2, vth=1, hard reset.
//   - Layer0 GEMM once (input is time-invariant across T).
//   - Per timestep: lif0 -> [WMMA GEMM1 fused with LIF1 epilogue] -> fused fc2.
//   - GEMM: v_wmma_f32_16x16x32_f16, 2x2 register blocking (32x32 per wave).
// ---------------------------------------------------------------------------

typedef __attribute__((ext_vector_type(16))) _Float16 v16h;
typedef __attribute__((ext_vector_type(8)))  _Float16 v8h;
typedef __attribute__((ext_vector_type(8)))  float    v8f;

#define B_      512
#define H_      1024
#define T_      32
#define K0PAD   800   // 784 padded up to a multiple of 32

// ---- f32 -> f16 convert with zero right-padding of each row ----------------
__global__ void cvt_pad_f16(const float* __restrict__ src, _Float16* __restrict__ dst,
                            int rows, int cin, int cout) {
  int i = blockIdx.x * blockDim.x + threadIdx.x;
  int total = rows * cout;
  if (i >= total) return;
  int r = i / cout, c = i % cout;
  dst[i] = (c < cin) ? (_Float16)src[(size_t)r * cin + c] : (_Float16)0.0f;
}

// ---- WMMA GEMM with 2x2 register blocking ----------------------------------
// C[M,N] = A[M,K](f16 row-major, lda) * Wt[N,K](f16 row-major, ldb)^T + bias
// Each wave owns a 32x32 output block (4 accumulator tiles). K % 32 == 0.
// FUSE_LIF: apply LIF membrane update in the epilogue and emit f16 spikes.
template <bool FUSE_LIF>
__global__ void gemm_wmma2x2(const _Float16* __restrict__ A, int lda,
                             const _Float16* __restrict__ Wt, int ldb,
                             const float* __restrict__ bias,
                             float* __restrict__ C,          // !FUSE_LIF
                             float* __restrict__ v,          // FUSE_LIF
                             _Float16* __restrict__ spike,   // FUSE_LIF
                             int N, int K, int totalWaves) {
  int waveId = blockIdx.x * (blockDim.x >> 5) + (threadIdx.x >> 5);
  if (waveId >= totalWaves) return;
  int wavesN = N >> 5;
  int wm = waveId / wavesN;
  int wn = waveId % wavesN;

  int lane = threadIdx.x & 31;
  int r16  = lane & 15;       // A row within tile; N column for B/C
  int hi   = lane >> 4;       // 0 or 1

  const _Float16* arow0 = A  + (size_t)(wm * 32 + r16) * lda + hi * 16;
  const _Float16* arow1 = arow0 + (size_t)16 * lda;
  const _Float16* brow0 = Wt + (size_t)(wn * 32 + r16) * ldb + hi * 16;
  const _Float16* brow1 = brow0 + (size_t)16 * ldb;

  v8f acc00 = {}, acc01 = {}, acc10 = {}, acc11 = {};

  for (int k = 0; k < K; k += 32) {
    // stream hints a few iterations ahead (speculative; safe past buffer end)
    __builtin_prefetch(arow0 + k + 256, 0, 1);
    __builtin_prefetch(arow1 + k + 256, 0, 1);
    __builtin_prefetch(brow0 + k + 256, 0, 1);
    __builtin_prefetch(brow1 + k + 256, 0, 1);

    // A fragments: elems 0..15 = K {k+hi*16..+15}
    v8h t0 = *(const v8h*)(arow0 + k);
    v8h t1 = *(const v8h*)(arow0 + k + 8);
    v8h t2 = *(const v8h*)(arow1 + k);
    v8h t3 = *(const v8h*)(arow1 + k + 8);
    v16h a0, a1;
#pragma unroll
    for (int i = 0; i < 8; ++i) {
      a0[i] = t0[i]; a0[8 + i] = t1[i];
      a1[i] = t2[i]; a1[8 + i] = t3[i];
    }
    // B fragments: 16 contiguous f16 along K of the weight row (32B aligned)
    v16h b0 = *(const v16h*)(brow0 + k);
    v16h b1 = *(const v16h*)(brow1 + k);

    acc00 = __builtin_amdgcn_wmma_f32_16x16x32_f16(false, a0, false, b0, (short)0, acc00, false, false);
    acc01 = __builtin_amdgcn_wmma_f32_16x16x32_f16(false, a0, false, b1, (short)0, acc01, false, false);
    acc10 = __builtin_amdgcn_wmma_f32_16x16x32_f16(false, a1, false, b0, (short)0, acc10, false, false);
    acc11 = __builtin_amdgcn_wmma_f32_16x16x32_f16(false, a1, false, b1, (short)0, acc11, false, false);
  }

  auto emit = [&](const v8f& acc, int mi, int ni) {
    int n = wn * 32 + ni * 16 + r16;
    float bv = bias[n];
#pragma unroll
    for (int r = 0; r < 8; ++r) {
      int m = wm * 32 + mi * 16 + hi * 8 + r;
      size_t idx = (size_t)m * N + n;
      float val = acc[r] + bv;
      if constexpr (FUSE_LIF) {
        float vv = v[idx];
        vv += (val - vv) * 0.5f;                  // tau=2, vreset=0
        float s = (vv >= 1.0f) ? 1.0f : 0.0f;     // heaviside(v - vth)
        v[idx] = (s != 0.0f) ? 0.0f : vv;         // hard reset
        spike[idx] = (_Float16)s;
      } else {
        C[idx] = val;
      }
    }
  };
  emit(acc00, 0, 0);
  emit(acc01, 0, 1);
  emit(acc10, 1, 0);
  emit(acc11, 1, 1);
}

// ---- One LIF step (layer 0; cur is time-invariant) -------------------------
__global__ void lif_step(const float* __restrict__ cur, float* __restrict__ v,
                         _Float16* __restrict__ spike, int n) {
  int i = blockIdx.x * blockDim.x + threadIdx.x;
  if (i >= n) return;
  float vv = v[i];
  vv += (cur[i] - vv) * 0.5f;
  float s = (vv >= 1.0f) ? 1.0f : 0.0f;
  v[i] = (s != 0.0f) ? 0.0f : vv;
  spike[i] = (_Float16)s;
}

// ---- Fused layer 2 (1024 -> 10): GEMV + LIF + spike-count accumulate -------
__global__ void fc2_lif_accum(const _Float16* __restrict__ s1,  // [512][1024]
                              const float* __restrict__ W2,     // [10][1024]
                              const float* __restrict__ b2,
                              float* __restrict__ v2,           // [512*10]
                              float* __restrict__ count) {      // [512*10] (d_out)
  int idx = blockIdx.x * blockDim.x + threadIdx.x;
  if (idx >= B_ * 10) return;
  int b = idx / 10, o = idx % 10;
  const _Float16* sr = s1 + (size_t)b * H_;
  const float*    wr = W2 + (size_t)o * H_;
  float acc = b2[o];
#pragma unroll 8
  for (int k = 0; k < H_; ++k) acc += (float)sr[k] * wr[k];
  float vv = v2[idx];
  vv += (acc - vv) * 0.5f;
  float s = (vv >= 1.0f) ? 1.0f : 0.0f;
  v2[idx] = (s != 0.0f) ? 0.0f : vv;
  count[idx] += s;
}

extern "C" void kernel_launch(void* const* d_in, const int* in_sizes, int n_in,
                              void* d_out, int out_size, void* d_ws, size_t ws_size,
                              hipStream_t stream) {
  const float* x  = (const float*)d_in[0];   // [512,784]
  const float* W0 = (const float*)d_in[1];   // [1024,784]
  const float* b0 = (const float*)d_in[2];   // [1024]
  const float* W1 = (const float*)d_in[3];   // [1024,1024]
  const float* b1 = (const float*)d_in[4];   // [1024]
  const float* W2 = (const float*)d_in[5];   // [10,1024]
  const float* b2 = (const float*)d_in[6];   // [10]
  float* count = (float*)d_out;              // [512,10]

  // ---- workspace carve-up (all offsets multiples of 256) ----
  char* ws = (char*)d_ws;
  size_t off = 0;
  auto alloc = [&](size_t bytes) { char* p = ws + off; off += (bytes + 255) & ~(size_t)255; return p; };
  _Float16* xh   = (_Float16*)alloc((size_t)B_ * K0PAD * 2);  // x padded to K=800
  _Float16* W0h  = (_Float16*)alloc((size_t)H_ * K0PAD * 2);  // W0 padded
  _Float16* W1h  = (_Float16*)alloc((size_t)H_ * H_ * 2);
  float*    cur0 = (float*)   alloc((size_t)B_ * H_ * 4);
  float*    v0   = (float*)   alloc((size_t)B_ * H_ * 4);
  float*    v1   = (float*)   alloc((size_t)B_ * H_ * 4);
  _Float16* s0   = (_Float16*)alloc((size_t)B_ * H_ * 2);
  _Float16* s1   = (_Float16*)alloc((size_t)B_ * H_ * 2);
  float*    v2   = (float*)   alloc((size_t)B_ * 10 * 4);
  (void)ws_size;

  // ---- zero persistent state each call (deterministic, capture-safe) ----
  hipMemsetAsync(v0, 0, (size_t)B_ * H_ * 4, stream);
  hipMemsetAsync(v1, 0, (size_t)B_ * H_ * 4, stream);
  hipMemsetAsync(v2, 0, (size_t)B_ * 10 * 4, stream);
  hipMemsetAsync(count, 0, (size_t)B_ * 10 * 4, stream);

  // ---- f32 -> f16 conversions (pad K 784 -> 800 with zeros) ----
  {
    int n = B_ * K0PAD;
    cvt_pad_f16<<<(n + 255) / 256, 256, 0, stream>>>(x, xh, B_, 784, K0PAD);
    n = H_ * K0PAD;
    cvt_pad_f16<<<(n + 255) / 256, 256, 0, stream>>>(W0, W0h, H_, 784, K0PAD);
    n = H_ * H_;
    cvt_pad_f16<<<(n + 255) / 256, 256, 0, stream>>>(W1, W1h, H_, H_, H_);
  }

  // ---- layer 0 GEMM (once — input identical for every timestep) ----
  const int waves01 = (B_ / 32) * (H_ / 32);  // 512 waves, 32x32 per wave
  gemm_wmma2x2<false><<<(waves01 + 3) / 4, 128, 0, stream>>>(
      xh, K0PAD, W0h, K0PAD, b0, cur0, nullptr, nullptr, H_, K0PAD, waves01);

  // ---- timestep loop: lif0 -> gemm1(+lif1 fused) -> fused fc2 ----
  const int nElem = B_ * H_;
  const int lifGrid = (nElem + 255) / 256;
  for (int t = 0; t < T_; ++t) {
    lif_step<<<lifGrid, 256, 0, stream>>>(cur0, v0, s0, nElem);
    gemm_wmma2x2<true><<<(waves01 + 3) / 4, 128, 0, stream>>>(
        s0, H_, W1h, H_, b1, nullptr, v1, s1, H_, H_, waves01);
    fc2_lif_accum<<<(B_ * 10 + 255) / 256, 256, 0, stream>>>(s1, W2, b2, v2, count);
  }
}
